// S4DMeanFlowNetV5_64175401337569
// MI455X (gfx1250) — matrix operand order repaired and checked by reading, not compiled, not run on hardware
//
#include <hip/hip_runtime.h>
#include <cstdint>
#include <cstddef>

// ---------------------------------------------------------------------------
// CDNA5 (gfx1250) S4D MeanFlow forward pass.
// All linears via v_wmma_f32_16x16x32_f16, f16-fed, 2x4 register blocking.
// Conv uses sliding register windows (4 l-values/thread, branch-free taps).
// ---------------------------------------------------------------------------

typedef __attribute__((ext_vector_type(16))) _Float16 v16h;
typedef __attribute__((ext_vector_type(8)))  _Float16 v8h;
typedef __attribute__((ext_vector_type(8)))  float    v8f;

#define DM   192     // d_model
#define DM2  384     // 2*d_model
#define NB   1024    // batch
#define LCTX 72
#define LPRD 24
#define EMB  256

enum { EPI_BIAS = 0, EPI_SILU = 1, EPI_MUL = 2, EPI_BIDIR = 3, EPI_RES = 4 };

// ---------------------------------------------------------------------------
// WMMA GEMM: epi( A[M,K](f16) * Wt[N,K](f16)^T + bias[N] ) -> C (f32) / Ch (f16)
// One 32x64 macro-tile per wave (2x4 grid of 16x16 WMMA tiles), 8 waves/block.
// ---------------------------------------------------------------------------
template <int EPI>
__global__ __launch_bounds__(256) void gemm_f16(
    const _Float16* __restrict__ A, const _Float16* __restrict__ Wt,
    const float* __restrict__ bias, float* __restrict__ C,
    _Float16* __restrict__ Ch, int M, int N, int K,
    const float* __restrict__ e1, const float* __restrict__ e2,
    const float* __restrict__ e3, int L)
{
  const int nmt   = N >> 6;               // macro tiles along N (64 wide)
  const int total = (M >> 5) * nmt;       // x macro tiles along M (32 tall)
  int t = blockIdx.x * 8 + (threadIdx.x >> 5);
  if (t >= total) return;
  const int mt   = t / nmt;
  const int nt   = t - mt * nmt;
  const int lane = threadIdx.x & 31;
  const int l15  = lane & 15;
  const int hi   = lane >> 4;
  const int kbA  = hi << 4;               // 0 or 16 (A frag K sub-offset)
  const int kbB  = hi << 4;               // 0 or 16 (B frag K sub-offset)

  const _Float16* Ar0 = A + (size_t)(mt * 32 + l15) * K;
  const _Float16* Ar1 = Ar0 + (size_t)16 * K;
  const _Float16* Wb  = Wt + (size_t)(nt * 64 + l15) * K;
  const size_t wstep  = (size_t)16 * K;

  v8f acc[2][4];
#pragma unroll
  for (int i = 0; i < 2; ++i)
#pragma unroll
    for (int j = 0; j < 4; ++j)
#pragma unroll
      for (int r = 0; r < 8; ++r) acc[i][j][r] = 0.f;

  for (int k0 = 0; k0 < K; k0 += 32) {
    v8h a0l = *(const v8h*)(Ar0 + k0 + kbA);
    v8h a0h = *(const v8h*)(Ar0 + k0 + 8 + kbA);
    v8h a1l = *(const v8h*)(Ar1 + k0 + kbA);
    v8h a1h = *(const v8h*)(Ar1 + k0 + 8 + kbA);
    v16h a0 = __builtin_shufflevector(a0l, a0h, 0,1,2,3,4,5,6,7,8,9,10,11,12,13,14,15);
    v16h a1 = __builtin_shufflevector(a1l, a1h, 0,1,2,3,4,5,6,7,8,9,10,11,12,13,14,15);
    v16h b0 = *(const v16h*)(Wb + 0 * wstep + k0 + kbB);
    v16h b1 = *(const v16h*)(Wb + 1 * wstep + k0 + kbB);
    v16h b2 = *(const v16h*)(Wb + 2 * wstep + k0 + kbB);
    v16h b3 = *(const v16h*)(Wb + 3 * wstep + k0 + kbB);
    acc[0][0] = __builtin_amdgcn_wmma_f32_16x16x32_f16(false, a0, false, b0, (short)0, acc[0][0], false, false);
    acc[0][1] = __builtin_amdgcn_wmma_f32_16x16x32_f16(false, a0, false, b1, (short)0, acc[0][1], false, false);
    acc[0][2] = __builtin_amdgcn_wmma_f32_16x16x32_f16(false, a0, false, b2, (short)0, acc[0][2], false, false);
    acc[0][3] = __builtin_amdgcn_wmma_f32_16x16x32_f16(false, a0, false, b3, (short)0, acc[0][3], false, false);
    acc[1][0] = __builtin_amdgcn_wmma_f32_16x16x32_f16(false, a1, false, b0, (short)0, acc[1][0], false, false);
    acc[1][1] = __builtin_amdgcn_wmma_f32_16x16x32_f16(false, a1, false, b1, (short)0, acc[1][1], false, false);
    acc[1][2] = __builtin_amdgcn_wmma_f32_16x16x32_f16(false, a1, false, b2, (short)0, acc[1][2], false, false);
    acc[1][3] = __builtin_amdgcn_wmma_f32_16x16x32_f16(false, a1, false, b3, (short)0, acc[1][3], false, false);
  }

  const int rb = hi << 3;
#pragma unroll
  for (int j = 0; j < 4; ++j) {
    const int   col = nt * 64 + j * 16 + l15;
    const float bn  = bias[col];
#pragma unroll
    for (int i = 0; i < 2; ++i) {
#pragma unroll
      for (int r = 0; r < 8; ++r) {
        int    m = mt * 32 + i * 16 + rb + r;
        size_t o = (size_t)m * N + col;
        float  v = acc[i][j][r] + bn;
        if (EPI == EPI_SILU) v = v / (1.f + expf(-v));
        if (EPI == EPI_MUL)  v *= e1[o];
        if (EPI == EPI_RES)  v += e1[o];
        if (EPI == EPI_BIDIR) {
          v += e1[(size_t)m * DM + col] * e2[col];     // + x*D   (N==DM)
          int bb = m / L;
          float sc = e3[(size_t)bb * DM2 + col];
          float sh = e3[(size_t)bb * DM2 + DM + col];
          v = v * (1.f + sc) + sh;                     // FiLM
        }
        if (C)  C[o]  = v;
        if (Ch) Ch[o] = (_Float16)v;
      }
    }
  }
}

// ---------------------------------------------------------------------------
// Weight convert: src[K,N] f32  ->  dst[N,Kp] f16 (zero-padded K)
// ---------------------------------------------------------------------------
__global__ void wconv_kernel(const float* __restrict__ src, _Float16* __restrict__ dst,
                             int K, int N, int Kp)
{
  int t = blockIdx.x * 256 + threadIdx.x;
  if (t >= N * Kp) return;
  int n = t / Kp, k = t - n * Kp;
  dst[t] = (k < K) ? (_Float16)src[(size_t)k * N + n] : (_Float16)0.f;
}

// ---------------------------------------------------------------------------
// LayerNorm over d=192: one wave per row. f32 output.
// ---------------------------------------------------------------------------
__global__ __launch_bounds__(256) void ln_kernel(
    const float* __restrict__ x, const float* __restrict__ g,
    const float* __restrict__ bb, float* __restrict__ y, int M)
{
  int row = blockIdx.x * 8 + (threadIdx.x >> 5);
  if (row >= M) return;
  int lane = threadIdx.x & 31;
  const float* xr = x + (size_t)row * DM;
  float v[6]; float s = 0.f;
#pragma unroll
  for (int i = 0; i < 6; ++i) { v[i] = xr[lane + 32 * i]; s += v[i]; }
  for (int m = 16; m > 0; m >>= 1) s += __shfl_xor(s, m, 32);
  float mu = s * (1.f / DM);
  float q = 0.f;
#pragma unroll
  for (int i = 0; i < 6; ++i) { float d = v[i] - mu; q += d * d; }
  for (int m = 16; m > 0; m >>= 1) q += __shfl_xor(q, m, 32);
  float inv = rsqrtf(q * (1.f / DM) + 1e-5f);
  float* yr = y + (size_t)row * DM;
#pragma unroll
  for (int i = 0; i < 6; ++i) {
    int c = lane + 32 * i;
    yr[c] = (v[i] - mu) * inv * g[c] + bb[c];
  }
}

// Same but f16 output (feeds q/k/v GEMMs directly).
__global__ __launch_bounds__(256) void ln16_kernel(
    const float* __restrict__ x, const float* __restrict__ g,
    const float* __restrict__ bb, _Float16* __restrict__ y, int M)
{
  int row = blockIdx.x * 8 + (threadIdx.x >> 5);
  if (row >= M) return;
  int lane = threadIdx.x & 31;
  const float* xr = x + (size_t)row * DM;
  float v[6]; float s = 0.f;
#pragma unroll
  for (int i = 0; i < 6; ++i) { v[i] = xr[lane + 32 * i]; s += v[i]; }
  for (int m = 16; m > 0; m >>= 1) s += __shfl_xor(s, m, 32);
  float mu = s * (1.f / DM);
  float q = 0.f;
#pragma unroll
  for (int i = 0; i < 6; ++i) { float d = v[i] - mu; q += d * d; }
  for (int m = 16; m > 0; m >>= 1) q += __shfl_xor(q, m, 32);
  float inv = rsqrtf(q * (1.f / DM) + 1e-5f);
  _Float16* yr = y + (size_t)row * DM;
#pragma unroll
  for (int i = 0; i < 6; ++i) {
    int c = lane + 32 * i;
    yr[c] = (_Float16)((v[i] - mu) * inv * g[c] + bb[c]);
  }
}

// ---------------------------------------------------------------------------
// S4D kernel generation -> Kt[l*192 + h] (transposed for coalesced conv)
// ---------------------------------------------------------------------------
__global__ void s4dk_kernel(const float* __restrict__ A_imag, const float* __restrict__ C_im,
                            const float* __restrict__ C_re, const float* __restrict__ log_A_real,
                            const float* __restrict__ log_dt, float* __restrict__ Kt, int L)
{
  int t = blockIdx.x * 256 + threadIdx.x;
  if (t >= DM * L) return;
  int hh = t / L, l = t - hh * L;
  float dt = expf(log_dt[hh]);
  float acc = 0.f;
  for (int n = 0; n < 32; ++n) {
    float Are = -expf(log_A_real[hh * 32 + n]);
    float Aim = A_imag[hh * 32 + n];
    float dre = Are * dt, dim = Aim * dt;
    float e  = expf(dre);
    float er = e * cosf(dim) - 1.f, ei = e * sinf(dim);      // exp(dtA)-1
    float cr = C_re[hh * 32 + n], ci = C_im[hh * 32 + n];
    float nr = cr * er - ci * ei, ni = cr * ei + ci * er;    // C*(exp(dtA)-1)
    float den = Are * Are + Aim * Aim;
    float c2r = (nr * Are + ni * Aim) / den;                 // / A
    float c2i = (ni * Are - nr * Aim) / den;
    float el = expf(dre * (float)l);
    float pr = el * cosf(dim * (float)l), pi = el * sinf(dim * (float)l);
    acc += c2r * pr - c2i * pi;
  }
  Kt[(size_t)l * DM + hh] = 2.f * acc;
}

// ---------------------------------------------------------------------------
// Bidirectional causal conv: x (B,L,192) f32 -> ycat (B,L,384) f16 [fwd|bwd]
// Each thread: fixed channel d, 4 consecutive l values, sliding register
// windows (zero-filled at bounds) -> 8 unconditional FMAs per tap,
// 2 K-loads + <=2 x-loads per tap. Grid = B * (L/4), blockDim = 192.
// ---------------------------------------------------------------------------
__global__ __launch_bounds__(192) void conv_kernel(
    const float* __restrict__ x, const float* __restrict__ Ktf,
    const float* __restrict__ Ktb, _Float16* __restrict__ ycat, int L)
{
  const int nch = L >> 2;                 // l-chunks per batch row
  int blk = blockIdx.x;
  int b   = blk / nch;
  int l0  = (blk - b * nch) << 2;
  int d   = threadIdx.x;
  const float* xb = x + (size_t)b * L * DM + d;

  // windows: wf[u] = x[l0+u-j] (0 if OOB), wb[u] = x[l0+u+j] (0 if OOB)
  float wf0, wf1, wf2, wf3, wb0, wb1, wb2, wb3;
  wf0 = wb0 = xb[(size_t)(l0 + 0) * DM];
  wf1 = wb1 = xb[(size_t)(l0 + 1) * DM];
  wf2 = wb2 = xb[(size_t)(l0 + 2) * DM];
  wf3 = wb3 = xb[(size_t)(l0 + 3) * DM];

  float af0 = 0.f, af1 = 0.f, af2 = 0.f, af3 = 0.f;
  float ab0 = 0.f, ab1 = 0.f, ab2 = 0.f, ab3 = 0.f;

  for (int j = 0; j < L; ++j) {
    float kf = Ktf[(size_t)j * DM + d];
    float kb = Ktb[(size_t)j * DM + d];
    af0 += kf * wf0; af1 += kf * wf1; af2 += kf * wf2; af3 += kf * wf3;
    ab0 += kb * wb0; ab1 += kb * wb1; ab2 += kb * wb2; ab3 += kb * wb3;
    // slide fwd window down: wf[u] <- wf[u-1]; wf0 <- x[l0-j-1] or 0
    wf3 = wf2; wf2 = wf1; wf1 = wf0;
    int fi = l0 - j - 1;
    wf0 = (fi >= 0) ? xb[(size_t)fi * DM] : 0.f;
    // slide bwd window up: wb[u] <- wb[u+1]; wb3 <- x[l0+4+j] or 0
    wb0 = wb1; wb1 = wb2; wb2 = wb3;
    int bi = l0 + 4 + j;
    wb3 = (bi < L) ? xb[(size_t)bi * DM] : 0.f;
  }

  _Float16* yr = ycat + (size_t)(b * L + l0) * DM2;
  yr[d]              = (_Float16)af0;  yr[DM + d]             = (_Float16)ab0;
  yr[DM2 + d]        = (_Float16)af1;  yr[DM2 + DM + d]       = (_Float16)ab1;
  yr[2 * DM2 + d]    = (_Float16)af2;  yr[2 * DM2 + DM + d]   = (_Float16)ab2;
  yr[3 * DM2 + d]    = (_Float16)af3;  yr[3 * DM2 + DM + d]   = (_Float16)ab3;
}

// ---------------------------------------------------------------------------
// sinusoidal embeddings (f16 out, feeds time_mlp1)
// ---------------------------------------------------------------------------
__global__ void emb_kernel(const float* __restrict__ t, const float* __restrict__ h,
                           _Float16* __restrict__ e0, int B_)
{
  int i = blockIdx.x * 256 + threadIdx.x;
  if (i >= B_ * 128) return;
  int b = i >> 7, c = i & 127;
  int grp = c >> 5, j = c & 31;
  float f  = expf(-(logf(10000.f) / 31.f) * (float)j);
  float xv = (grp < 2) ? t[b] : h[b];
  float a  = xv * f;
  e0[i] = (_Float16)(((grp & 1) == 0) ? sinf(a) : cosf(a));
}

// silu f32 -> f16 (feeds film GEMM)
__global__ void silu16_kernel(const float* __restrict__ x, _Float16* __restrict__ y, int n)
{
  int t = blockIdx.x * 256 + threadIdx.x;
  if (t >= n) return;
  float v = x[t];
  y[t] = (_Float16)(v / (1.f + expf(-v)));
}

// ctx_input: (B,11,Lc) + ch-0 nonzero mask -> transposed & zero-padded f16 A (B*Lc, 32)
__global__ void ctxin_kernel(const float* __restrict__ ctx, _Float16* __restrict__ Apad, int L)
{
  int t = blockIdx.x * 256 + threadIdx.x;
  if (t >= NB * L * 32) return;
  int c = t & 31, bl = t >> 5;
  int b = bl / L, l = bl - b * L;
  float v = 0.f;
  if (c < 11)       v = ctx[((size_t)b * 11 + c) * L + l];
  else if (c == 11) v = (fabsf(ctx[(size_t)b * 11 * L + l]) > 1e-6f) ? 1.f : 0.f;
  Apad[t] = (_Float16)v;
}

__global__ void predin_kernel(const float* __restrict__ noisy, _Float16* __restrict__ Apad, int Mp)
{
  int t = blockIdx.x * 256 + threadIdx.x;
  if (t >= Mp * 32) return;
  int c = t & 31, m = t >> 5;
  Apad[t] = (_Float16)((c == 0) ? noisy[m] : 0.f);
}

// mean over L: x (B,L,192) f32 -> out (B,192) f16 (feeds ctx_pool GEMM)
__global__ void mean_kernel(const float* __restrict__ x, _Float16* __restrict__ out, int L)
{
  int t = blockIdx.x * 256 + threadIdx.x;
  if (t >= NB * DM) return;
  int b = t / DM, d = t - b * DM;
  const float* xb = x + (size_t)b * L * DM + d;
  float s = 0.f;
  for (int l = 0; l < L; ++l) s += xb[(size_t)l * DM];
  out[t] = (_Float16)(s * (1.f / (float)L));
}

// ---------------------------------------------------------------------------
// Cross-attention: one (batch, head) per block. Lq=24, Lc=72, dh=48.
// Q/K/V f32 in LDS, output f16 (feeds o-proj GEMM).
// ---------------------------------------------------------------------------
__global__ __launch_bounds__(256) void attn_kernel(
    const float* __restrict__ Q, const float* __restrict__ Kc,
    const float* __restrict__ Vc, _Float16* __restrict__ O)
{
  __shared__ float Qs[24 * 48];
  __shared__ float Ks[72 * 48];
  __shared__ float Vs[72 * 48];
  __shared__ float S[24 * 72];
  int bh = blockIdx.x;
  int b = bh >> 2, hd = bh & 3;
  int tid = threadIdx.x;
  const float* Qb = Q  + (size_t)b * LPRD * DM + hd * 48;
  const float* Kb = Kc + (size_t)b * LCTX * DM + hd * 48;
  const float* Vb = Vc + (size_t)b * LCTX * DM + hd * 48;
  for (int t = tid; t < 24 * 48; t += 256) {
    int q = t / 48, i = t - q * 48;
    Qs[t] = Qb[(size_t)q * DM + i];
  }
  for (int t = tid; t < 72 * 48; t += 256) {
    int k = t / 48, i = t - k * 48;
    Ks[t] = Kb[(size_t)k * DM + i];
    Vs[t] = Vb[(size_t)k * DM + i];
  }
  __syncthreads();
  const float scale = rsqrtf(48.f);
  for (int t = tid; t < 24 * 72; t += 256) {
    int q = t / 72, k = t - q * 72;
    float s = 0.f;
#pragma unroll
    for (int i = 0; i < 48; ++i) s += Qs[q * 48 + i] * Ks[k * 48 + i];
    S[t] = s * scale;
  }
  __syncthreads();
  int wave = tid >> 5, lane = tid & 31;
  for (int r = wave; r < 24; r += 8) {
    float vals[3];
    float mx = -1e30f;
#pragma unroll
    for (int i = 0; i < 3; ++i) {
      int c = lane + 32 * i;
      vals[i] = (c < 72) ? S[r * 72 + c] : -1e30f;
      mx = fmaxf(mx, vals[i]);
    }
    for (int m = 16; m > 0; m >>= 1) mx = fmaxf(mx, __shfl_xor(mx, m, 32));
    float s = 0.f;
#pragma unroll
    for (int i = 0; i < 3; ++i) {
      int c = lane + 32 * i;
      if (c < 72) { vals[i] = expf(vals[i] - mx); s += vals[i]; }
    }
    for (int m = 16; m > 0; m >>= 1) s += __shfl_xor(s, m, 32);
    float inv = 1.f / s;
#pragma unroll
    for (int i = 0; i < 3; ++i) {
      int c = lane + 32 * i;
      if (c < 72) S[r * 72 + c] = vals[i] * inv;
    }
  }
  __syncthreads();
  _Float16* Ob = O + (size_t)b * LPRD * DM + hd * 48;
  for (int t = tid; t < 24 * 48; t += 256) {
    int q = t / 48, i = t - q * 48;
    float o = 0.f;
    for (int k = 0; k < 72; ++k) o += S[q * 72 + k] * Vs[k * 48 + i];
    Ob[(size_t)q * DM + i] = (_Float16)o;
  }
}

// ---------------------------------------------------------------------------
// Output head: LN -> SiLU -> dot with out_proj.W (192,1) + b. One wave/row.
// ---------------------------------------------------------------------------
__global__ __launch_bounds__(256) void out_kernel(
    const float* __restrict__ x, const float* __restrict__ g,
    const float* __restrict__ bb, const float* __restrict__ Wo,
    const float* __restrict__ bo, float* __restrict__ out, int M)
{
  int row = blockIdx.x * 8 + (threadIdx.x >> 5);
  if (row >= M) return;
  int lane = threadIdx.x & 31;
  const float* xr = x + (size_t)row * DM;
  float v[6]; float s = 0.f;
#pragma unroll
  for (int i = 0; i < 6; ++i) { v[i] = xr[lane + 32 * i]; s += v[i]; }
  for (int m = 16; m > 0; m >>= 1) s += __shfl_xor(s, m, 32);
  float mu = s * (1.f / DM);
  float q = 0.f;
#pragma unroll
  for (int i = 0; i < 6; ++i) { float d = v[i] - mu; q += d * d; }
  for (int m = 16; m > 0; m >>= 1) q += __shfl_xor(q, m, 32);
  float inv = rsqrtf(q * (1.f / DM) + 1e-5f);
  float acc = 0.f;
#pragma unroll
  for (int i = 0; i < 6; ++i) {
    int c = lane + 32 * i;
    float n = (v[i] - mu) * inv * g[c] + bb[c];
    acc += (n / (1.f + expf(-n))) * Wo[c];
  }
  for (int m = 16; m > 0; m >>= 1) acc += __shfl_xor(acc, m, 32);
  if (lane == 0) out[row] = acc + bo[0];
}

// ===========================================================================
// Host orchestration
// ===========================================================================
static inline int gemmGrid(int M, int N) {
  int tiles = (M / 32) * (N / 64);
  return (tiles + 7) / 8;
}

template <int E>
static void rungemm(hipStream_t s, const _Float16* A, const _Float16* Wt, const float* b,
                    float* C, _Float16* Ch, int M, int N, int K,
                    const float* e1 = nullptr, const float* e2 = nullptr,
                    const float* e3 = nullptr, int L = 1)
{
  gemm_f16<E><<<gemmGrid(M, N), 256, 0, s>>>(A, Wt, b, C, Ch, M, N, K, e1, e2, e3, L);
}

struct BlockP {
  const float *D, *bidirW, *bidirb, *downW, *downb, *filmW, *filmb, *gateW, *gateb;
  const float *kb_Aim, *kb_Cim, *kb_Cre, *kb_lAr, *kb_ldt;
  const float *kf_Aim, *kf_Cim, *kf_Cre, *kf_lAr, *kf_ldt;
  const float *lnb, *lng, *upW, *upb;
};
struct BlockW { _Float16 *bidir, *down, *film, *gate, *up; };

extern "C" void kernel_launch(void* const* d_in, const int* in_sizes, int n_in,
                              void* d_out, int out_size, void* d_ws, size_t ws_size,
                              hipStream_t stream)
{
  (void)in_sizes; (void)out_size; (void)ws_size;
  const int Mc = NB * LCTX, Mp = NB * LPRD;

  const float* noisy = (const float*)d_in[0];
  const float* t_in  = (const float*)d_in[1];
  const float* h_in  = (const float*)d_in[2];
  const float* ctxin = (const float*)d_in[3];

  // Param leaves in jax pytree (sorted-key) order.
  const float* P[257] = {nullptr};
  for (int i = 0; i < 257 && (4 + i) < n_in; ++i) P[i] = (const float*)d_in[4 + i];

  // --- workspace bump allocator (deterministic layout each call) -----------
  uintptr_t base = (uintptr_t)d_ws;
  size_t off = 0;
  auto allocF = [&](size_t n) -> float* {
    float* p = (float*)(base + off); off += ((n * 4 + 255) / 256) * 256; return p;
  };
  auto allocH = [&](size_t n) -> _Float16* {
    _Float16* p = (_Float16*)(base + off); off += ((n * 2 + 255) / 256) * 256; return p;
  };

  // --- weight conversion to f16 transposed [N, Kp] -------------------------
  auto wconv = [&](const float* src, int K, int Kp, int N) -> _Float16* {
    _Float16* dst = allocH((size_t)N * Kp);
    int tot = N * Kp;
    wconv_kernel<<<(tot + 255) / 256, 256, 0, stream>>>(src, dst, K, N, Kp);
    return dst;
  };

  auto fillBlock = [&](int bse, BlockP& bp) {
    bp.D = P[bse + 0];  bp.bidirW = P[bse + 1]; bp.bidirb = P[bse + 2];
    bp.downW = P[bse + 3]; bp.downb = P[bse + 4];
    bp.filmW = P[bse + 5]; bp.filmb = P[bse + 6];
    bp.gateW = P[bse + 7]; bp.gateb = P[bse + 8];
    bp.kb_Aim = P[bse + 9];  bp.kb_Cim = P[bse + 10]; bp.kb_Cre = P[bse + 11];
    bp.kb_lAr = P[bse + 12]; bp.kb_ldt = P[bse + 13];
    bp.kf_Aim = P[bse + 14]; bp.kf_Cim = P[bse + 15]; bp.kf_Cre = P[bse + 16];
    bp.kf_lAr = P[bse + 17]; bp.kf_ldt = P[bse + 18];
    bp.lnb = P[bse + 19]; bp.lng = P[bse + 20];
    bp.upW = P[bse + 21]; bp.upb = P[bse + 22];
  };

  BlockP cbp[3], pbp[6];
  BlockW cbw[3], pbw[6];
  for (int i = 0; i < 3; ++i) fillBlock(36 + 23 * i, cbp[i]);
  for (int i = 0; i < 6; ++i) fillBlock(113 + 23 * i, pbp[i]);
  for (int i = 0; i < 3; ++i) {
    cbw[i].bidir = wconv(cbp[i].bidirW, DM2, DM2, DM);
    cbw[i].down  = wconv(cbp[i].downW,  DM2, DM2, DM);
    cbw[i].film  = wconv(cbp[i].filmW,  EMB, EMB, DM2);
    cbw[i].gate  = wconv(cbp[i].gateW,  DM,  DM,  DM2);
    cbw[i].up    = wconv(cbp[i].upW,    DM,  DM,  DM2);
  }
  for (int i = 0; i < 6; ++i) {
    pbw[i].bidir = wconv(pbp[i].bidirW, DM2, DM2, DM);
    pbw[i].down  = wconv(pbp[i].downW,  DM2, DM2, DM);
    pbw[i].film  = wconv(pbp[i].filmW,  EMB, EMB, DM2);
    pbw[i].gate  = wconv(pbp[i].gateW,  DM,  DM,  DM2);
    pbw[i].up    = wconv(pbp[i].upW,    DM,  DM,  DM2);
  }
  _Float16 *aqW[3], *akW[3], *avW[3], *aoW[3];
  for (int a = 0; a < 3; ++a) {
    int bse = a * 12;
    akW[a] = wconv(P[bse + 0], DM, DM, DM);
    aoW[a] = wconv(P[bse + 6], DM, DM, DM);
    aqW[a] = wconv(P[bse + 8], DM, DM, DM);
    avW[a] = wconv(P[bse + 10], DM, DM, DM);
  }
  _Float16* poolWt  = wconv(P[105], DM, DM, EMB);       // ctx_pool (192,256)
  _Float16* cprojWt = wconv(P[107], 12, 32, DM);        // ctx_proj (12,192) pad K->32
  _Float16* pprojWt = wconv(P[251], 1, 32, DM);         // pred_proj (1,192) pad K->32
  _Float16* mlp1Wt  = wconv(P[253], 128, 128, EMB);     // time_mlp1
  _Float16* mlp2Wt  = wconv(P[255], EMB, EMB, EMB);     // time_mlp2

  // --- activation buffers --------------------------------------------------
  float*    ctxRes  = allocF((size_t)Mc * DM);
  float*    predRes = allocF((size_t)Mp * DM);
  float*    bufX    = allocF((size_t)Mc * DM);     // LN out (f32, conv + x*D)
  float*    bufG    = allocF((size_t)Mc * DM2);    // gate out (f32, MUL epi)
  float*    bufFilm = allocF((size_t)NB * DM2);
  float*    Ktf     = allocF((size_t)LCTX * DM);
  float*    Ktb     = allocF((size_t)LCTX * DM);
  float*    embA    = allocF((size_t)NB * EMB);
  float*    embB    = allocF((size_t)NB * EMB);
  float*    Qp      = allocF((size_t)Mp * DM);
  float*    Kc      = allocF((size_t)Mc * DM);
  float*    Vc      = allocF((size_t)Mc * DM);
  _Float16* ycat16  = allocH((size_t)Mc * DM2);    // conv out -> bidir A
  _Float16* bufZ16  = allocH((size_t)Mc * DM);     // bidir out -> gate/up A
  _Float16* bufU16  = allocH((size_t)Mc * DM2);    // up out -> down A
  _Float16* bufX16  = allocH((size_t)Mc * DM);     // attn LN out -> q/k/v A
  _Float16* attnO16 = allocH((size_t)Mp * DM);     // attn out -> o-proj A
  _Float16* emb016  = allocH((size_t)NB * 128);
  _Float16* embT16  = allocH((size_t)NB * EMB);
  _Float16* sembA16 = allocH((size_t)NB * EMB);
  _Float16* sembB16 = allocH((size_t)NB * EMB);
  _Float16* meanC16 = allocH((size_t)NB * DM);
  _Float16* ApadC16 = allocH((size_t)Mc * 32);
  _Float16* ApadP16 = allocH((size_t)Mp * 32);

  // --- time embedding ------------------------------------------------------
  emb_kernel<<<(NB * 128 + 255) / 256, 256, 0, stream>>>(t_in, h_in, emb016, NB);
  rungemm<EPI_SILU>(stream, emb016, mlp1Wt, P[254], nullptr, embT16, NB, EMB, 128);
  rungemm<EPI_BIAS>(stream, embT16, mlp2Wt, P[256], embA, nullptr, NB, EMB, EMB);
  silu16_kernel<<<(NB * EMB + 255) / 256, 256, 0, stream>>>(embA, sembA16, NB * EMB);

  // --- context projection --------------------------------------------------
  ctxin_kernel<<<(NB * LCTX * 32 + 255) / 256, 256, 0, stream>>>(ctxin, ApadC16, LCTX);
  rungemm<EPI_BIAS>(stream, ApadC16, cprojWt, P[108], ctxRes, nullptr, Mc, DM, 32);

  // --- s4d block runner ----------------------------------------------------
  auto s4dBlock = [&](const BlockP& bp, const BlockW& bw, float* res, int M, int L,
                      const _Float16* semb) {
    rungemm<EPI_BIAS>(stream, semb, bw.film, bp.filmb, bufFilm, nullptr, NB, DM2, EMB);
    s4dk_kernel<<<(DM * L + 255) / 256, 256, 0, stream>>>(
        bp.kf_Aim, bp.kf_Cim, bp.kf_Cre, bp.kf_lAr, bp.kf_ldt, Ktf, L);
    s4dk_kernel<<<(DM * L + 255) / 256, 256, 0, stream>>>(
        bp.kb_Aim, bp.kb_Cim, bp.kb_Cre, bp.kb_lAr, bp.kb_ldt, Ktb, L);
    ln_kernel<<<(M + 7) / 8, 256, 0, stream>>>(res, bp.lng, bp.lnb, bufX, M);
    conv_kernel<<<M / 4, 192, 0, stream>>>(bufX, Ktf, Ktb, ycat16, L);
    rungemm<EPI_BIDIR>(stream, ycat16, bw.bidir, bp.bidirb, nullptr, bufZ16, M, DM, DM2,
                       bufX, bp.D, bufFilm, L);
    rungemm<EPI_SILU>(stream, bufZ16, bw.gate, bp.gateb, bufG, nullptr, M, DM2, DM);
    rungemm<EPI_MUL>(stream, bufZ16, bw.up, bp.upb, nullptr, bufU16, M, DM2, DM, bufG);
    rungemm<EPI_RES>(stream, bufU16, bw.down, bp.downb, res, nullptr, M, DM, DM2, res);
  };

  for (int i = 0; i < 3; ++i) s4dBlock(cbp[i], cbw[i], ctxRes, Mc, LCTX, sembA16);

  // --- context pooling into embedding --------------------------------------
  mean_kernel<<<(NB * DM + 255) / 256, 256, 0, stream>>>(ctxRes, meanC16, LCTX);
  rungemm<EPI_RES>(stream, meanC16, poolWt, P[106], embB, nullptr, NB, EMB, DM, embA);
  silu16_kernel<<<(NB * EMB + 255) / 256, 256, 0, stream>>>(embB, sembB16, NB * EMB);

  // --- prediction branch ---------------------------------------------------
  predin_kernel<<<(Mp * 32 + 255) / 256, 256, 0, stream>>>(noisy, ApadP16, Mp);
  rungemm<EPI_BIAS>(stream, ApadP16, pprojWt, P[252], predRes, nullptr, Mp, DM, 32);

  int ai = 0;
  for (int i = 0; i < 6; ++i) {
    s4dBlock(pbp[i], pbw[i], predRes, Mp, LPRD, sembB16);
    if ((i & 1) == 1 && ai < 3) {
      int bse = ai * 12;
      const float *kb_ = P[bse + 1], *lnkvb = P[bse + 2], *lnkvg = P[bse + 3];
      const float *lnqb = P[bse + 4], *lnqg = P[bse + 5], *ob_ = P[bse + 7];
      const float *qb_ = P[bse + 9], *vb_ = P[bse + 11];
      ln16_kernel<<<(Mp + 7) / 8, 256, 0, stream>>>(predRes, lnqg, lnqb, bufX16, Mp);
      rungemm<EPI_BIAS>(stream, bufX16, aqW[ai], qb_, Qp, nullptr, Mp, DM, DM);
      ln16_kernel<<<(Mc + 7) / 8, 256, 0, stream>>>(ctxRes, lnkvg, lnkvb, bufX16, Mc);
      rungemm<EPI_BIAS>(stream, bufX16, akW[ai], kb_, Kc, nullptr, Mc, DM, DM);
      rungemm<EPI_BIAS>(stream, bufX16, avW[ai], vb_, Vc, nullptr, Mc, DM, DM);
      attn_kernel<<<NB * 4, 256, 0, stream>>>(Qp, Kc, Vc, attnO16);
      rungemm<EPI_RES>(stream, attnO16, aoW[ai], ob_, predRes, nullptr, Mp, DM, DM, predRes);
      ++ai;
    }
  }

  // --- output head ---------------------------------------------------------
  out_kernel<<<(Mp + 7) / 8, 256, 0, stream>>>(predRes, P[110], P[109], P[111],
                                               P[112], (float*)d_out, Mp);
}
